// GraphSAGE_21311627723552
// MI455X (gfx1250) — matrix operand order repaired and checked by reading, not compiled, not run on hardware
//
#include <hip/hip_runtime.h>

typedef __attribute__((ext_vector_type(16))) __bf16 v16bf;
typedef __attribute__((ext_vector_type(8)))  float  v8f;

// ---------------------------------------------------------------------------
// Zero fill (grid-stride)
// ---------------------------------------------------------------------------
__global__ void zero_f32_kernel(float* __restrict__ p, long long n) {
    long long i = (long long)blockIdx.x * blockDim.x + threadIdx.x;
    long long stride = (long long)gridDim.x * blockDim.x;
    for (; i < n; i += stride) p[i] = 0.0f;
}

// ---------------------------------------------------------------------------
// Edge-parallel mean-aggregation scatter: one wave (32 lanes) per edge.
// agg[dst] += x[src]; deg[dst] += 1 (deg pass only on layer 1).
// Atomics lower to global_atomic_add_f32; x / agg are L2-resident (192MB L2).
// ---------------------------------------------------------------------------
__global__ void scatter_add_kernel(const float* __restrict__ x,
                                   const long long* __restrict__ src,
                                   const long long* __restrict__ dst,
                                   float* __restrict__ agg,
                                   float* __restrict__ deg,
                                   int E, int D) {
    int gid  = blockIdx.x * blockDim.x + threadIdx.x;
    int e    = gid >> 5;
    int lane = gid & 31;
    if (e >= E) return;
    long long s = src[e];
    long long d = dst[e];
    const float* xs = x + s * (long long)D;
    float* ad = agg + d * (long long)D;
    for (int j = lane; j < D; j += 32) atomicAdd(&ad[j], xs[j]);
    if (deg != nullptr && lane == 0) atomicAdd(&deg[d], 1.0f);
}

// ---------------------------------------------------------------------------
// agg[i][j] /= max(deg[i], 1)   (isolated nodes -> 0)
// ---------------------------------------------------------------------------
__global__ void divide_deg_kernel(float* __restrict__ agg,
                                  const float* __restrict__ deg,
                                  int total, int D) {
    int i = blockIdx.x * blockDim.x + threadIdx.x;
    if (i >= total) return;
    int row = i / D;
    agg[i] = agg[i] / fmaxf(deg[row], 1.0f);
}

// ---------------------------------------------------------------------------
// Fragment builders: f32 row-major source -> bf16 WMMA fragment, using
// B128 vector loads (all offsets 16B-aligned: K steps are multiples of 8
// floats, row strides multiples of 32 floats).
//
// A 16x32 bf16 (CDNA5 wave32): lane m (&15), half = lane>>4:
//   a[i]   = A[m][k0 + half*8 + i]        i in 0..7
//   a[i+8] = A[m][k0 + 16 + half*8 + i]
// B 32x16 bf16: lane n (&15), half = lane>>4:
//   b[i]   = W[n][k0 + half*8 + i]        i in 0..7    (B[k][n] = W[n][k])
//   b[i+8] = W[n][k0 + 16 + half*8 + i]
// ---------------------------------------------------------------------------
__device__ __forceinline__ v16bf load_fragA(const float* __restrict__ base,
                                            int ld, int row, int kk, int half) {
    const float* p = base + (size_t)row * ld + kk + half * 8;
    float4 f0 = *(const float4*)(p);
    float4 f1 = *(const float4*)(p + 4);
    float4 f2 = *(const float4*)(p + 16);
    float4 f3 = *(const float4*)(p + 20);
    v16bf a;
    a[0]  = (__bf16)f0.x; a[1]  = (__bf16)f0.y; a[2]  = (__bf16)f0.z; a[3]  = (__bf16)f0.w;
    a[4]  = (__bf16)f1.x; a[5]  = (__bf16)f1.y; a[6]  = (__bf16)f1.z; a[7]  = (__bf16)f1.w;
    a[8]  = (__bf16)f2.x; a[9]  = (__bf16)f2.y; a[10] = (__bf16)f2.z; a[11] = (__bf16)f2.w;
    a[12] = (__bf16)f3.x; a[13] = (__bf16)f3.y; a[14] = (__bf16)f3.z; a[15] = (__bf16)f3.w;
    return a;
}

__device__ __forceinline__ v16bf load_fragB(const float* __restrict__ base,
                                            int ld, int wrow, int kk, int half) {
    const float* p = base + (size_t)wrow * ld + kk + half * 8;
    float4 f0 = *(const float4*)(p);
    float4 f1 = *(const float4*)(p + 4);
    float4 f2 = *(const float4*)(p + 16);
    float4 f3 = *(const float4*)(p + 20);
    v16bf b;
    b[0]  = (__bf16)f0.x; b[1]  = (__bf16)f0.y; b[2]  = (__bf16)f0.z; b[3]  = (__bf16)f0.w;
    b[4]  = (__bf16)f1.x; b[5]  = (__bf16)f1.y; b[6]  = (__bf16)f1.z; b[7]  = (__bf16)f1.w;
    b[8]  = (__bf16)f2.x; b[9]  = (__bf16)f2.y; b[10] = (__bf16)f2.z; b[11] = (__bf16)f2.w;
    b[12] = (__bf16)f3.x; b[13] = (__bf16)f3.y; b[14] = (__bf16)f3.z; b[15] = (__bf16)f3.w;
    return b;
}

// ---------------------------------------------------------------------------
// Fused SAGE GEMM:  out = [Aagg | Aroot] @ [Bagg | Broot]^T + bias (+ ReLU)
//   Aagg/Aroot: M x Khalf (row-major, f32)   (mean-aggregated / root features)
//   Bagg/Broot: Nout x Khalf (row-major, f32) (lin_l / lin_r weights)
// One wave computes a 32x32 output tile: 2 A-frags x 2 B-frags -> 4
// v_wmma_f32_16x16x32_bf16 per K-step (register blocking halves fragment
// loads per WMMA in each direction).
// C/D 16x16 f32: element (m = r + 8*half, n = lane&15) in acc[r].
// ---------------------------------------------------------------------------
__global__ void __launch_bounds__(256)
sage_gemm_kernel(const float* __restrict__ Aagg, const float* __restrict__ Aroot,
                 const float* __restrict__ Bagg, const float* __restrict__ Broot,
                 const float* __restrict__ bias, float* __restrict__ out,
                 int M, int Khalf, int Nout, int relu) {
    int lane = threadIdx.x & 31;
    int wave = (blockIdx.x * blockDim.x + threadIdx.x) >> 5;
    int ntn   = Nout >> 5;                    // 32-wide col tiles
    int tiles = (M >> 5) * ntn;               // 32-tall row tiles
    if (wave >= tiles) return;                // wave-uniform: EXEC stays all-1s

    int rt = wave / ntn;
    int ct = wave - rt * ntn;
    int row0 = rt << 5;
    int col0 = ct << 5;
    int half = lane >> 4;
    int l15  = lane & 15;

    v8f acc00 = {}, acc01 = {}, acc10 = {}, acc11 = {};
    int Ktot = Khalf << 1;
    for (int k0 = 0; k0 < Ktot; k0 += 32) {
        const float* pa;
        const float* pb;
        int kk;
        if (k0 < Khalf) { pa = Aagg;  pb = Bagg;  kk = k0; }
        else            { pa = Aroot; pb = Broot; kk = k0 - Khalf; }

        v16bf a0 = load_fragA(pa, Khalf, row0 + l15,      kk, half);
        v16bf a1 = load_fragA(pa, Khalf, row0 + 16 + l15, kk, half);
        v16bf b0 = load_fragB(pb, Khalf, col0 + l15,      kk, half);
        v16bf b1 = load_fragB(pb, Khalf, col0 + 16 + l15, kk, half);

        acc00 = __builtin_amdgcn_wmma_f32_16x16x32_bf16(false, a0, false, b0,
                                                        (short)0, acc00, false, false);
        acc01 = __builtin_amdgcn_wmma_f32_16x16x32_bf16(false, a0, false, b1,
                                                        (short)0, acc01, false, false);
        acc10 = __builtin_amdgcn_wmma_f32_16x16x32_bf16(false, a1, false, b0,
                                                        (short)0, acc10, false, false);
        acc11 = __builtin_amdgcn_wmma_f32_16x16x32_bf16(false, a1, false, b1,
                                                        (short)0, acc11, false, false);
    }

    int   n0  = col0 + l15;
    int   n1  = col0 + 16 + l15;
    float bv0 = bias[n0];
    float bv1 = bias[n1];
    int   m0  = row0 + (half << 3);           // rows for acc0x
    int   m1  = row0 + 16 + (half << 3);      // rows for acc1x
#pragma unroll
    for (int r = 0; r < 8; ++r) {
        float v00 = acc00[r] + bv0;
        float v01 = acc01[r] + bv1;
        float v10 = acc10[r] + bv0;
        float v11 = acc11[r] + bv1;
        if (relu) {
            v00 = fmaxf(v00, 0.0f); v01 = fmaxf(v01, 0.0f);
            v10 = fmaxf(v10, 0.0f); v11 = fmaxf(v11, 0.0f);
        }
        out[(size_t)(m0 + r) * Nout + n0] = v00;
        out[(size_t)(m0 + r) * Nout + n1] = v01;
        out[(size_t)(m1 + r) * Nout + n0] = v10;
        out[(size_t)(m1 + r) * Nout + n1] = v11;
    }
}

// ---------------------------------------------------------------------------
// In-place row log_softmax over 64 columns. One wave32 per row:
// lane owns cols {lane, lane+32}; reductions via __shfl_xor.
// ---------------------------------------------------------------------------
__global__ void log_softmax64_kernel(float* __restrict__ out, int M) {
    int lane = threadIdx.x & 31;
    int row  = (blockIdx.x * blockDim.x + threadIdx.x) >> 5;
    if (row >= M) return;
    float* r = out + (size_t)row * 64;
    float v0 = r[lane];
    float v1 = r[lane + 32];
    float m = fmaxf(v0, v1);
#pragma unroll
    for (int o = 16; o > 0; o >>= 1) m = fmaxf(m, __shfl_xor(m, o, 32));
    float s = expf(v0 - m) + expf(v1 - m);
#pragma unroll
    for (int o = 16; o > 0; o >>= 1) s += __shfl_xor(s, o, 32);
    float lg = m + logf(s);
    r[lane]      = v0 - lg;
    r[lane + 32] = v1 - lg;
}

// ---------------------------------------------------------------------------
// Host launcher
// ---------------------------------------------------------------------------
extern "C" void kernel_launch(void* const* d_in, const int* in_sizes, int n_in,
                              void* d_out, int out_size, void* d_ws, size_t ws_size,
                              hipStream_t stream) {
    const float*     x       = (const float*)d_in[0];
    const long long* eidx    = (const long long*)d_in[1];   // int64 per reference
    const float*     w1_agg  = (const float*)d_in[2];
    const float*     b1      = (const float*)d_in[3];
    const float*     w1_root = (const float*)d_in[4];
    const float*     w2_agg  = (const float*)d_in[5];
    const float*     b2      = (const float*)d_in[6];
    const float*     w2_root = (const float*)d_in[7];
    float*           out     = (float*)d_out;

    const int d_hid = in_sizes[3];                 // 256
    const int d_out_dim = in_sizes[6];             // 64
    const int d_in_dim  = in_sizes[2] / d_hid;     // 128
    const int N = in_sizes[0] / d_in_dim;          // 100000
    const int E = in_sizes[1] / 2;                 // 640000
    const long long* src = eidx;
    const long long* dst = eidx + E;

    // Workspace layout (f32): deg[N] | agg1[N*d_in] | h1[N*d_hid] | agg2[N*d_hid]
    float* ws   = (float*)d_ws;
    float* deg  = ws;
    float* agg1 = deg  + N;
    float* h1   = agg1 + (size_t)N * d_in_dim;
    float* agg2 = h1   + (size_t)N * d_hid;

    const int TB = 256;

    // ---- Layer 1 ----
    long long z1 = (long long)N * (1 + d_in_dim);            // deg + agg1 contiguous
    zero_f32_kernel<<<4096, TB, 0, stream>>>(deg, z1);

    int scat_blocks = (E * 32 + TB - 1) / TB;
    scatter_add_kernel<<<scat_blocks, TB, 0, stream>>>(x, src, dst, agg1, deg, E, d_in_dim);

    int tot1 = N * d_in_dim;
    divide_deg_kernel<<<(tot1 + TB - 1) / TB, TB, 0, stream>>>(agg1, deg, tot1, d_in_dim);

    int tiles1  = (N >> 5) * (d_hid >> 5);                   // 32x32 tiles
    int blocks1 = (tiles1 + 7) / 8;                          // 8 waves / block
    sage_gemm_kernel<<<blocks1, TB, 0, stream>>>(agg1, x, w1_agg, w1_root, b1, h1,
                                                 N, d_in_dim, d_hid, /*relu=*/1);

    // ---- Layer 2 ----
    zero_f32_kernel<<<4096, TB, 0, stream>>>(agg2, (long long)N * d_hid);

    scatter_add_kernel<<<scat_blocks, TB, 0, stream>>>(h1, src, dst, agg2, nullptr, E, d_hid);

    int tot2 = N * d_hid;
    divide_deg_kernel<<<(tot2 + TB - 1) / TB, TB, 0, stream>>>(agg2, deg, tot2, d_hid);

    int tiles2  = (N >> 5) * (d_out_dim >> 5);
    int blocks2 = (tiles2 + 7) / 8;
    sage_gemm_kernel<<<blocks2, TB, 0, stream>>>(agg2, h1, w2_agg, w2_root, b2, out,
                                                 N, d_hid, d_out_dim, /*relu=*/0);

    // ---- log_softmax over 64 columns ----
    int ls_blocks = (N * 32 + TB - 1) / TB;
    log_softmax64_kernel<<<ls_blocks, TB, 0, stream>>>(out, N);
}
